// GCN_Agent_13108240187703
// MI455X (gfx1250) — matrix operand order repaired and checked by reading, not compiled, not run on hardware
//
#include <hip/hip_runtime.h>
#include <stdint.h>

// ---- problem constants ----
#define B_SZ    1024
#define N_NODES 256
#define H_DIM   64
#define A_DIM   256
#define K_ITERS 4
#define M_ROWS  (B_SZ * N_NODES)   // 262144

typedef unsigned short u16;
typedef __attribute__((ext_vector_type(16))) __bf16 bf16x16;
typedef __attribute__((ext_vector_type(8)))  float  f32x8;

union FragAB { bf16x16 bf; uint4 q[2]; };

__device__ __forceinline__ u16 f2bf(float f) {
  union { float f; unsigned u; } c; c.f = f;
  unsigned r = c.u + 0x7FFFu + ((c.u >> 16) & 1u);   // round-to-nearest-even
  return (u16)(r >> 16);
}
__device__ __forceinline__ float bf2f(u16 h) {
  union { unsigned u; float f; } c; c.u = ((unsigned)h) << 16;
  return c.f;
}

// ---- W staging into LDS as W^T (64 cols x 32 K, +8 pad), bf16 ----
// Uniform "full tile" fast path: no per-lane predication in the hot loop.
__device__ __forceinline__ void stage_w(const float* __restrict__ Wp, int ldW, int col0,
                                        int k0, int KW, int tid, u16* Bs) {
  const bool full = (k0 + 32 <= KW);      // uniform -> scalar branch
#pragma unroll
  for (int it = 0; it < 2; ++it) {
    const int u  = tid + it * 256;        // 512 units: float4 along n
    const int k  = u >> 4, n4 = (u & 15) << 2;
    const float* src = Wp + (long)(k0 + k) * ldW + col0 + n4;
    float4 v;
    if (full) {
      v = *(const float4*)src;
    } else {
      v = make_float4(0.f, 0.f, 0.f, 0.f);
      if (k0 + k < KW) v = *(const float4*)src;
    }
    u16* bp = Bs + n4 * 40 + k;
    bp[0]   = f2bf(v.x);
    bp[40]  = f2bf(v.y);
    bp[80]  = f2bf(v.z);
    bp[120] = f2bf(v.w);
  }
}
// bf16 operand (msg in the agg GEMM): always full tiles (K=256).
__device__ __forceinline__ void stage_w(const u16* __restrict__ Wp, int ldW, int col0,
                                        int k0, int KW, int tid, u16* Bs) {
  const bool full = (k0 + 32 <= KW);      // uniform (always true for agg)
  const int k  = tid >> 3, n8 = (tid & 7) << 3;   // 256 units: 8 bf16 along n
  const u16* src = Wp + (long)(k0 + k) * ldW + col0 + n8;
  uint4 v = {0u, 0u, 0u, 0u};
  if (full) {
    v = *(const uint4*)src;
  } else if (k0 + k < KW) {
    v = *(const uint4*)src;
  }
  const u16* pv = (const u16*)&v;
  u16* bp = Bs + n8 * 40 + k;
#pragma unroll
  for (int j = 0; j < 8; ++j) bp[j * 40] = pv[j];
}

__device__ __forceinline__ void store_out(float* p, float v) { *p = v; }
__device__ __forceinline__ void store_out(u16* p, float v)   { *p = f2bf(v); }

// -----------------------------------------------------------------------------
// Tiled WMMA GEMM:
//   Out[bat](m, col0+[0..64)) = act( A1[bat] @ W1  (+ A2 @ W2) (+ E @ WE) + bias )
// A operands: bf16 in global, fragments loaded DIRECTLY (two global_load_b128
// per lane per k-step; no LDS, no conversion). W: staged to LDS as W^T bf16.
// Block = 256 threads (8 wave32). Tile = 128 rows x 64 cols.
// Each wave: 16 rows -> 4 accumulators of v_wmma_f32_16x16x32_bf16.
// K must be a multiple of 32; KW = #valid W rows (zero-pad beyond).
// -----------------------------------------------------------------------------
template <typename TW, typename TO>
__global__ __launch_bounds__(256)
void wmma_gemm_kernel(
    const u16* __restrict__ A1, int ldA1, long bsA1, int K1,
    const TW* __restrict__ W1, int ldW1, int KW1,
    const u16* __restrict__ A2, int ldA2, int K2,
    const TW* __restrict__ W2, int ldW2,
    long bsW,
    const float* __restrict__ bias,
    const float* __restrict__ E, const float* __restrict__ WE, int ldWE,
    TO* __restrict__ Out, int ldOut, long bsOut,
    int act_tanh)
{
  __shared__ __align__(16) u16 Bs[64 * 40];

  const int tid  = threadIdx.x;
  const int lane = tid & 31;
  const int wv   = tid >> 5;          // wave 0..7
  const int hb   = (lane >> 4) & 1;   // wave half
  const int ln   = lane & 15;
  const int m0   = blockIdx.x * 128;
  const long bat = blockIdx.y;
  const int col0 = blockIdx.z * 64;

  f32x8 acc[4];
#pragma unroll
  for (int t = 0; t < 4; ++t)
#pragma unroll
    for (int v = 0; v < 8; ++v) acc[t][v] = 0.0f;

  for (int seg = 0; seg < 2; ++seg) {
    const u16* Ap; int ldA, K, KW; const TW* Wp;
    if (seg == 0) {
      Ap = A1 + bat * bsA1; ldA = ldA1; Wp = W1 + bat * bsW; K = K1; KW = KW1;
    } else {
      if (A2 == nullptr) break;
      Ap = A2; ldA = ldA2; Wp = W2; K = K2; KW = K2;
    }
    const int ldW = (seg == 0) ? ldW1 : ldW2;
    // per-lane A row base (16B-aligned: ldA multiple of 8 bf16)
    const u16* arow = Ap + (long)(m0 + wv * 16 + ln) * ldA;

    for (int k0 = 0; k0 < K; k0 += 32) {
      stage_w(Wp, ldW, col0, k0, KW, tid, Bs);
      __syncthreads();

      // A fragment direct from global: K runs [8*hb,+8) and [16+8*hb,+8)
      FragAB fa;
      fa.q[0] = *(const uint4*)(arow + k0 + 8 * hb);
      fa.q[1] = *(const uint4*)(arow + k0 + 16 + 8 * hb);

#pragma unroll
      for (int t = 0; t < 4; ++t) {
        FragAB fb;
        const u16* bp = &Bs[(t * 16 + ln) * 40 + 8 * hb];
        fb.q[0] = *(const uint4*)(bp);
        fb.q[1] = *(const uint4*)(bp + 16);
        acc[t] = __builtin_amdgcn_wmma_f32_16x16x32_bf16(
            false, fa.bf, false, fb.bf, (short)0, acc[t], false, false);
      }
      __syncthreads();
    }
  }

  // epilogue: C/D layout -> row = v + 8*hb, col = ln (within 16-col tile)
  TO* outp = Out + bat * bsOut;
  const int rbase = m0 + wv * 16 + hb * 8;
#pragma unroll
  for (int t = 0; t < 4; ++t) {
    const int col = col0 + t * 16 + ln;
#pragma unroll
    for (int v = 0; v < 8; ++v) {
      const long row = rbase + v;
      float r = acc[t][v];
      if (E != nullptr) {
#pragma unroll
        for (int j = 0; j < 8; ++j)
          r += E[row * 8 + j] * WE[j * ldWE + col];
      }
      if (bias != nullptr) r += bias[col];
      if (act_tanh) r = tanhf(r);
      store_out(&outp[row * (long)ldOut + col], r);
    }
  }
}

// ---- feats: x[:, :-1].reshape(B,N,16) -> bf16, K-padded to 32 with zeros ----
__global__ void feats_kernel(const float* __restrict__ x, u16* __restrict__ feats) {
  long i = (long)blockIdx.x * 256 + threadIdx.x;       // M_ROWS*32 total
  long r = i >> 5; int c = (int)(i & 31);
  long b = r >> 8; int n = (int)(r & 255);
  float v = (c < 16) ? x[b * 4097 + n * 16 + c] : 0.0f;
  feats[i] = f2bf(v);
}

// ---- An[i][j] = adj_d[j][i] / deg_i  (unit diag, transposed, row-normalized), bf16
__global__ void adjprep_kernel(const float* __restrict__ adj, u16* __restrict__ An) {
  int i = threadIdx.x;   // one block of 256
  float deg = 0.0f;
  for (int j = 0; j < 256; ++j)
    deg += (j == i) ? 1.0f : adj[j * 256 + i];
  float inv = 1.0f / deg;
  for (int j = 0; j < 256; ++j) {
    float a = (j == i) ? 1.0f : adj[j * 256 + i];
    An[i * 256 + j] = f2bf(a * inv);
  }
}

// ---- e = tanh(x[:, -1:] @ Wex + bex), shape (B, 8), f32 ----
__global__ void extras_kernel(const float* __restrict__ x, const float* __restrict__ W,
                              const float* __restrict__ b, float* __restrict__ E) {
  int i = blockIdx.x * 256 + threadIdx.x;   // 8192
  if (i < B_SZ * 8) {
    int bb = i >> 3, j = i & 7;
    E[i] = tanhf(x[(long)bb * 4097 + 4096] * W[j] + b[j]);
  }
}

__global__ void tanh_bf_kernel(u16* __restrict__ p, long n) {
  long i = (long)blockIdx.x * 256 + threadIdx.x;
  if (i < n) p[i] = f2bf(tanhf(bf2f(p[i])));
}

// ---- critic lin2: value[b] = z[b] . W + bias ----
__global__ void critic_head_kernel(const u16* __restrict__ z, const float* __restrict__ W,
                                   const float* __restrict__ b, float* __restrict__ val) {
  int r = blockIdx.x * 256 + threadIdx.x;
  if (r < B_SZ) {
    float s = b[0];
#pragma unroll
    for (int k = 0; k < 64; ++k) s += bf2f(z[(long)r * 64 + k]) * W[k];
    val[r] = s;
  }
}

// ---- log-softmax + gather + entropy; one wave32 per row of 256 logits ----
__global__ __launch_bounds__(256)
void policy_kernel(const float* __restrict__ logits, const int* __restrict__ action,
                   float* __restrict__ out) {
  int lane = threadIdx.x & 31;
  int wv   = threadIdx.x >> 5;
  int row  = blockIdx.x * 8 + wv;          // 128 blocks x 8 rows
  const float* l = logits + (long)row * 256;
  float v[8];
  float mx = -3.402823466e38f;
#pragma unroll
  for (int i = 0; i < 8; ++i) { v[i] = l[lane + 32 * i]; mx = fmaxf(mx, v[i]); }
#pragma unroll
  for (int s = 16; s > 0; s >>= 1) mx = fmaxf(mx, __shfl_xor(mx, s, 32));
  float sum = 0.0f;
#pragma unroll
  for (int i = 0; i < 8; ++i) sum += __expf(v[i] - mx);
#pragma unroll
  for (int s = 16; s > 0; s >>= 1) sum += __shfl_xor(sum, s, 32);
  float logZ = mx + __logf(sum);
  float ent = 0.0f;
#pragma unroll
  for (int i = 0; i < 8; ++i) { float lp = v[i] - logZ; ent -= __expf(lp) * lp; }
#pragma unroll
  for (int s = 16; s > 0; s >>= 1) ent += __shfl_xor(ent, s, 32);
  if (lane == 0) {
    int a = action[row];
    out[row]        = (float)a;        // action (as float output)
    out[1024 + row] = l[a] - logZ;     // log p(a)
    out[2048 + row] = ent;             // entropy
  }
}

// -----------------------------------------------------------------------------
// host-side orchestration
// -----------------------------------------------------------------------------
struct HeadParams {
  const float *encW1, *encb1, *msgW1, *msgb1, *updW1, *updb1;
  const float *encW2, *encb2, *msgW2, *msgb2, *updW2, *updb2;
  const float *lin1W, *lin1b, *lin2W, *lin2b;
};

template <typename TW, typename TO>
static inline void gemm(hipStream_t s, dim3 grid,
    const u16* A1, int ldA1, long bsA1, int K1,
    const TW* W1, int ldW1, int KW1,
    const u16* A2, int ldA2, int K2, const TW* W2, int ldW2,
    long bsW, const float* bias,
    const float* E, const float* WE, int ldWE,
    TO* Out, int ldOut, long bsOut, int act)
{
  wmma_gemm_kernel<TW, TO><<<grid, dim3(256), 0, s>>>(
      A1, ldA1, bsA1, K1, W1, ldW1, KW1, A2, ldA2, K2, W2, ldW2,
      bsW, bias, E, WE, ldWE, Out, ldOut, bsOut, act);
}

static void run_gcn(hipStream_t s, const u16* in, int ldIn, int Kin, int KWin,
                    const float* encW, const float* encb,
                    const float* msgW, const float* msgb,
                    const float* updW, const float* updb,
                    const u16* An, u16* enc, u16* msg, u16* agg)
{
  dim3 gM(M_ROWS / 128, 1, 1);
  // enc = in @ encW + b
  gemm<float, u16>(s, gM, in, ldIn, 0, Kin, encW, 64, KWin,
                   nullptr, 0, 0, nullptr, 0, 0, encb,
                   nullptr, nullptr, 0, enc, 64, 0, 0);
  for (int it = 0; it < K_ITERS; ++it) {
    // msg = enc @ msgW + b
    gemm<float, u16>(s, gM, enc, 64, 0, 64, msgW, 64, 64,
                     nullptr, 0, 0, nullptr, 0, 0, msgb,
                     nullptr, nullptr, 0, msg, 64, 0, 0);
    // agg[b] = An @ msg[b]  (batched over B; An shared, W operand batched bf16)
    gemm<u16, u16>(s, dim3(N_NODES / 128, B_SZ, 1), An, 256, 0, 256, msg, 64, 256,
                   nullptr, 0, 0, nullptr, 0, (long)N_NODES * 64, nullptr,
                   nullptr, nullptr, 0, agg, 64, (long)N_NODES * 64, 0);
    // enc = tanh(enc @ updW[0:64] + agg @ updW[64:128] + b)  (row-local, in-place safe)
    gemm<float, u16>(s, gM, enc, 64, 0, 64, updW, 64, 64,
                     agg, 64, 64, updW + 64 * 64, 64, 0, updb,
                     nullptr, nullptr, 0, enc, 64, 0, 1);
  }
}

static void run_head_common(hipStream_t s, const HeadParams& hp, const u16* featsC,
                            const u16* An, const float* E,
                            u16* enc, u16* msg, u16* agg, u16* z)
{
  run_gcn(s, featsC, 32, 32, 16, hp.encW1, hp.encb1, hp.msgW1, hp.msgb1,
          hp.updW1, hp.updb1, An, enc, msg, agg);
  tanh_bf_kernel<<<65536, 256, 0, s>>>(enc, (long)M_ROWS * 64);
  run_gcn(s, enc, 64, 64, 64, hp.encW2, hp.encb2, hp.msgW2, hp.msgb2,
          hp.updW2, hp.updb2, An, enc, msg, agg);
  tanh_bf_kernel<<<65536, 256, 0, s>>>(enc, (long)M_ROWS * 64);
  // z = tanh(flat @ lin1W[0:16384] + E @ lin1W[16384:16392] + b)
  gemm<float, u16>(s, dim3(B_SZ / 128, 1, 1), enc, 16384, 0, 16384, hp.lin1W, 64, 16384,
                   nullptr, 0, 0, nullptr, 0, 0, hp.lin1b,
                   E, hp.lin1W + (long)16384 * 64, 64, z, 64, 0, 1);
}

static HeadParams unpack_head(void* const* d_in, int base) {
  HeadParams hp;
  // pytree (sorted-key) order per head: extra{W,b}, gcn1{enc,msg,upd}{W,b},
  // gcn2{...}, lin1{W,b}, lin2{W,b}
  hp.encW1 = (const float*)d_in[base + 2];  hp.encb1 = (const float*)d_in[base + 3];
  hp.msgW1 = (const float*)d_in[base + 4];  hp.msgb1 = (const float*)d_in[base + 5];
  hp.updW1 = (const float*)d_in[base + 6];  hp.updb1 = (const float*)d_in[base + 7];
  hp.encW2 = (const float*)d_in[base + 8];  hp.encb2 = (const float*)d_in[base + 9];
  hp.msgW2 = (const float*)d_in[base + 10]; hp.msgb2 = (const float*)d_in[base + 11];
  hp.updW2 = (const float*)d_in[base + 12]; hp.updb2 = (const float*)d_in[base + 13];
  hp.lin1W = (const float*)d_in[base + 14]; hp.lin1b = (const float*)d_in[base + 15];
  hp.lin2W = (const float*)d_in[base + 16]; hp.lin2b = (const float*)d_in[base + 17];
  return hp;
}

extern "C" void kernel_launch(void* const* d_in, const int* in_sizes, int n_in,
                              void* d_out, int out_size, void* d_ws, size_t ws_size,
                              hipStream_t stream) {
  const float* x      = (const float*)d_in[0];
  const int*   action = (const int*)d_in[1];
  const float* adj    = (const float*)d_in[2];
  HeadParams actor  = unpack_head(d_in, 3);
  HeadParams critic = unpack_head(d_in, 21);
  // reference bug replicated: BOTH heads use params['critic']['extra']
  const float* exW = (const float*)d_in[21];
  const float* exb = (const float*)d_in[22];

  // workspace layout (bf16 intermediates; iteration working set is L2-resident)
  char* ws = (char*)d_ws;
  u16*   featsC = (u16*)ws;                 ws += (size_t)M_ROWS * 32 * 2;   // 16.8 MB
  u16*   An     = (u16*)ws;                 ws += 256 * 256 * 2;
  float* Ebuf   = (float*)ws;               ws += (size_t)B_SZ * 8 * 4;
  u16*   enc    = (u16*)ws;                 ws += (size_t)M_ROWS * 64 * 2;   // 33.5 MB
  u16*   msg    = (u16*)ws;                 ws += (size_t)M_ROWS * 64 * 2;
  u16*   agg    = (u16*)ws;                 ws += (size_t)M_ROWS * 64 * 2;
  u16*   z      = (u16*)ws;                 ws += (size_t)B_SZ * 64 * 2;
  float* logits = (float*)ws;               ws += (size_t)B_SZ * A_DIM * 4;

  float* out = (float*)d_out;               // [action | lp_a | ent | value], 4x1024

  // prep
  feats_kernel<<<32768, 256, 0, stream>>>(x, featsC);
  adjprep_kernel<<<1, 256, 0, stream>>>(adj, An);
  extras_kernel<<<32, 256, 0, stream>>>(x, exW, exb, Ebuf);

  // actor head -> logits (f32)
  run_head_common(stream, actor, featsC, An, Ebuf, enc, msg, agg, z);
  gemm<float, float>(stream, dim3(B_SZ / 128, 1, A_DIM / 64), z, 64, 0, 64,
                     actor.lin2W, A_DIM, 64, nullptr, 0, 0, nullptr, 0, 0,
                     actor.lin2b, nullptr, nullptr, 0, logits, A_DIM, 0, 0);

  // critic head -> value (into out[3072:4096])
  run_head_common(stream, critic, featsC, An, Ebuf, enc, msg, agg, z);
  critic_head_kernel<<<4, 256, 0, stream>>>(z, critic.lin2W, critic.lin2b, out + 3072);

  // log-softmax / gather / entropy
  policy_kernel<<<128, 256, 0, stream>>>(logits, action, out);
}
